// SpectralAttentionLayer_7679401525511
// MI455X (gfx1250) — matrix operand order repaired and checked by reading, not compiled, not run on hardware
//
#include <hip/hip_runtime.h>
#include <hip/hip_bf16.h>

typedef __attribute__((ext_vector_type(16))) _Float16 v16h;
typedef __attribute__((ext_vector_type(8)))  float    v8f;
typedef __attribute__((ext_vector_type(4)))  int      v4i;

#define D_MODEL 256
#define KEIG    32
#define NHEAD   8
#define DHEAD   32
#define NBANDS  3

#if defined(__gfx1250__) && __has_builtin(__builtin_amdgcn_global_load_async_to_lds_b128)
#define HAVE_ASYNC_LDS 1
typedef __attribute__((address_space(1))) v4i gv4i;   // global int4
typedef __attribute__((address_space(3))) v4i lv4i;   // LDS int4
#endif

// ---------------------------------------------------------------------------
// Kernel 0: zero the accumulation workspace (XF and S are atomically reduced)
// ---------------------------------------------------------------------------
__global__ __launch_bounds__(256)
void k_init_ws(float* __restrict__ p, int n) {
  for (int i = blockIdx.x * blockDim.x + threadIdx.x; i < n;
       i += gridDim.x * blockDim.x)
    p[i] = 0.0f;
}

// ---------------------------------------------------------------------------
// Kernel 1 (fold): XF[b, k, d] = sum_n E[n,k] * x[b,n,d]   (32 x 256 per batch)
//                  S[k]        = sum_n E[n,k]              (for the bias term)
// WMMA: A = E^T tile (16 k x 32 n, f16), B = x tile (32 n x 16 d, f16),
//       C/D = f32 accumulators. Double-buffered fragment LDS: global loads for
//       chunk i+1 are issued (as one batched clause) before chunk i's WMMAs.
// ---------------------------------------------------------------------------
__global__ __launch_bounds__(256)
void k_fold(const float* __restrict__ x, const float* __restrict__ E,
            float* __restrict__ XF, float* __restrict__ S, int N) {
  // Fragments stored in lane/element order so reads are contiguous b128 DS ops.
  __shared__ __align__(32) _Float16 xb[2][16][32][16];  // 32 KB
  __shared__ __align__(32) _Float16 eb[2][2][32][16];   //  4 KB
  __shared__ float sloc[KEIG];

  const int b    = blockIdx.y;
  const int tid  = threadIdx.x;
  const int lane = tid & 31;
  const int wave = tid >> 5;
  const int kh   = wave >> 2;         // k-half this wave owns (0 or 1)
  const int dt0  = (wave & 3) * 4;    // first of 4 d-tiles this wave owns
  const int stride = gridDim.x * 32;

  const size_t xbase = (size_t)b * N * D_MODEL;
  float sreg[4] = {0.f, 0.f, 0.f, 0.f};   // per-thread partial column sums of E

  auto stage = [&](int base, int buf) {
    // Prefetch the chunk after this one (speculative; dropped if OOB).
    if (tid < 32 && base + stride + tid < N)
      __builtin_prefetch(x + xbase + (size_t)(base + stride + tid) * D_MODEL,
                         0, 1);
    // ---- phase 1: batched global loads (one clause, one wait) ----
    float4 vx[8];
#pragma unroll
    for (int i = 0; i < 8; ++i) {
      int idx = tid + i * 256;          // float4 slot 0..2047
      int r   = idx >> 6;               // local row 0..31
      int d0  = (idx & 63) << 2;        // column
      int n   = base + r;
      vx[i] = (n < N)
          ? *(const float4*)(x + xbase + (size_t)n * D_MODEL + d0)
          : make_float4(0.f, 0.f, 0.f, 0.f);
    }
    const int re = tid >> 3;            // E local row 0..31 (K of A)
    const int k0 = (tid & 7) << 2;      // eig index (M of A)
    const int ne = base + re;
    float4 vE = (ne < N)
        ? *(const float4*)(E + (size_t)ne * KEIG + k0)
        : make_float4(0.f, 0.f, 0.f, 0.f);

    // ---- phase 2: convert + scatter into fragment LDS ----
#pragma unroll
    for (int i = 0; i < 8; ++i) {
      int idx = tid + i * 256;
      int r   = idx >> 6;
      int d0  = (idx & 63) << 2;
      const float vv[4] = {vx[i].x, vx[i].y, vx[i].z, vx[i].w};
#pragma unroll
      for (int j = 0; j < 4; ++j) {
        int d = d0 + j;
        // B layout: lanes 0-15 hold K=0..15 (elem=K), lanes 16-31 K=16..31.
        xb[buf][d >> 4][(d & 15) + (r & 16)][r & 15] = (_Float16)vv[j];
      }
    }
    {
      const float vv[4] = {vE.x, vE.y, vE.z, vE.w};
#pragma unroll
      for (int j = 0; j < 4; ++j) {
        int k   = k0 + j;
        int khh = k >> 4, m = k & 15;
        // A k order this code uses: lanes0-15 K=0..15 (elem=K), lanes16-31 K=16..31
        int ln = m + ((re & 16) ? 16 : 0);
        int e  = re & 15;
        eb[buf][khh][ln][e] = (_Float16)vv[j];
        sreg[j] += vv[j];               // zero-padded rows contribute 0
      }
    }
  };

  if (tid < KEIG) sloc[tid] = 0.0f;
  __syncthreads();

  v8f acc[4] = {};
  int chunk = blockIdx.x * 32;
  int p = 0;
  if (chunk < N) stage(chunk, 0);
  __syncthreads();

  for (; chunk < N; chunk += stride) {
    int next = chunk + stride;
    if (next < N) stage(next, p ^ 1);   // issue next chunk's loads first

    v16h a = *(const v16h*)&eb[p][kh][lane][0];
#pragma unroll
    for (int j = 0; j < 4; ++j) {
      v16h bf = *(const v16h*)&xb[p][dt0 + j][lane][0];
      acc[j] = __builtin_amdgcn_wmma_f32_16x16x32_f16(
          false, a, false, bf, (short)0, acc[j], false, false);
    }
    __syncthreads();
    p ^= 1;
  }

  // ---- reduce partial tiles into XF via global f32 atomics ----
#pragma unroll
  for (int j = 0; j < 4; ++j) {
    int dbase = (dt0 + j) << 4;
#pragma unroll
    for (int v = 0; v < 8; ++v) {
      int m  = v + ((lane & 16) ? 8 : 0);       // C layout: M = vgpr (+8 hi half)
      int kk = (kh << 4) + m;
      int d  = dbase + (lane & 15);             // N = lane
      atomicAdd(&XF[((size_t)b * KEIG + kk) * D_MODEL + d], acc[j][v]);
    }
  }
  // ---- column sums of E: registers -> LDS -> global (once per block) ----
  {
    int k0 = (tid & 7) << 2;
#pragma unroll
    for (int j = 0; j < 4; ++j) atomicAdd(&sloc[k0 + j], sreg[j]);
  }
  __syncthreads();
  if (tid < KEIG) atomicAdd(&S[tid], sloc[tid]);
}

// ---------------------------------------------------------------------------
// Kernel 2 (spectral core): per (b,h) block.
//   Qf/Kf/Vf = XF @ W^T + S*bias, band-filter gain, 32x32 attention + softmax,
//   OF[b,k,h*32+d] = out_freq.  Tiny: pure VALU.
// ---------------------------------------------------------------------------
__global__ __launch_bounds__(256)
void k_spectral_attn(const float* __restrict__ XF, const float* __restrict__ S,
                     const float* __restrict__ ev,
                     const float* __restrict__ Wq, const float* __restrict__ bq,
                     const float* __restrict__ Wk, const float* __restrict__ bk,
                     const float* __restrict__ Wv, const float* __restrict__ bv,
                     const float* __restrict__ bb, const float* __restrict__ fw,
                     float* __restrict__ OF) {
  __shared__ float xf[KEIG][D_MODEL];     // 32 KB
  __shared__ float wr[DHEAD][D_MODEL];    // 32 KB (reused per matrix)
  __shared__ float qf[KEIG][DHEAD];
  __shared__ float kf[KEIG][DHEAD];
  __shared__ float vf[KEIG][DHEAD];
  __shared__ float g[KEIG], sl[KEIG];

  const int b = blockIdx.x >> 3;
  const int h = blockIdx.x & 7;
  const int tid = threadIdx.x;

  for (int i = tid; i < KEIG * D_MODEL / 4; i += 256)
    ((float4*)&xf[0][0])[i] =
        ((const float4*)(XF + (size_t)b * KEIG * D_MODEL))[i];

  if (tid < KEIG) {
    sl[tid] = S[tid];
    float emin = ev[0], emax = ev[KEIG - 1];       // eigenvalues are sorted
    float lam = (ev[tid] - emin) / (emax - emin + 1e-8f);
    float r = 0.0f;
    for (int i = 0; i < NBANDS; ++i) {
      float lo = bb[h * (NBANDS + 1) + i];
      float hi = bb[h * (NBANDS + 1) + i + 1];
      if (lam >= lo && lam < hi) r = fw[(h * NBANDS + i) * KEIG + tid];
    }
    g[tid] = r;
  }
  __syncthreads();

  auto do_mtx = [&](const float* __restrict__ W, const float* __restrict__ bias,
                    float (*dst)[DHEAD]) {
    for (int i = tid; i < DHEAD * D_MODEL / 4; i += 256)
      ((float4*)&wr[0][0])[i] =
          ((const float4*)(W + (size_t)h * DHEAD * D_MODEL))[i];
    __syncthreads();
    int k = tid >> 3, d0 = (tid & 7) << 2;
    float acc[4] = {0.f, 0.f, 0.f, 0.f};
    for (int c = 0; c < D_MODEL; ++c) {
      float xv = xf[k][c];
#pragma unroll
      for (int j = 0; j < 4; ++j) acc[j] = fmaf(xv, wr[d0 + j][c], acc[j]);
    }
#pragma unroll
    for (int j = 0; j < 4; ++j)
      dst[k][d0 + j] =
          (acc[j] + sl[k] * bias[h * DHEAD + d0 + j]) * g[k];
    __syncthreads();
  };
  do_mtx(Wq, bq, qf);
  do_mtx(Wk, bk, kf);
  do_mtx(Wv, bv, vf);

  if (tid < KEIG) {               // one wave does the 32x32 attention
    const float scale = 0.17677669529663687f;   // 1/sqrt(Dh)
    float sc[KEIG];
    float m = -3.4e38f;
    for (int l = 0; l < KEIG; ++l) {
      float s = 0.0f;
      for (int d = 0; d < DHEAD; ++d) s = fmaf(qf[tid][d], kf[l][d], s);
      s *= scale;
      sc[l] = s;
      m = fmaxf(m, s);
    }
    float sum = 0.0f;
    for (int l = 0; l < KEIG; ++l) { sc[l] = __expf(sc[l] - m); sum += sc[l]; }
    float inv = 1.0f / sum;
    float o[DHEAD];
    for (int d = 0; d < DHEAD; ++d) o[d] = 0.0f;
    for (int l = 0; l < KEIG; ++l) {
      float a = sc[l] * inv;
      for (int d = 0; d < DHEAD; ++d) o[d] = fmaf(a, vf[l][d], o[d]);
    }
    for (int d = 0; d < DHEAD; ++d)
      OF[((size_t)b * KEIG + tid) * D_MODEL + h * DHEAD + d] = o[d];
  }
}

// ---------------------------------------------------------------------------
// Kernel 3 (unfold): out[b,n,d] = sum_k E[n,k] * OF[b,k,d]
// K_EIG == 32 == WMMA K: one v_wmma_f32_16x16x32_f16 per 16x16 output tile.
// OF[b] (32 KB) is staged via async global->LDS copies (ASYNCcnt) when the
// toolchain exposes the gfx1250 builtin, then converted to f16 B-fragments.
// ---------------------------------------------------------------------------
__global__ __launch_bounds__(256)
void k_unfold(const float* __restrict__ OF, const float* __restrict__ E,
              float* __restrict__ out, int N) {
  __shared__ __align__(32) _Float16 ofb[16][32][16];  // B-fragments, 16 KB
#ifdef HAVE_ASYNC_LDS
  __shared__ __align__(16) float ofraw[KEIG][D_MODEL];  // raw staging, 32 KB
#endif

  const int b = blockIdx.y, tid = threadIdx.x;
  const int lane = tid & 31, wave = tid >> 5;

#ifdef HAVE_ASYNC_LDS
  {
    const float* src = OF + (size_t)b * KEIG * D_MODEL;
    // 256 lanes x 8 x b128: 32 KB fire-and-forget copy tracked by ASYNCcnt.
#pragma unroll
    for (int i = 0; i < 8; ++i) {
      int fi = (tid + i * 256) * 4;     // float index, 16B granules
      __builtin_amdgcn_global_load_async_to_lds_b128(
          (gv4i*)(src + fi),
          (lv4i*)(&ofraw[0][0] + fi),
          0, 0);
    }
#if __has_builtin(__builtin_amdgcn_s_wait_asynccnt)
    __builtin_amdgcn_s_wait_asynccnt(0);
#else
    asm volatile("s_wait_asynccnt 0x0" ::: "memory");
#endif
  }
  __syncthreads();
  // convert raw f32 -> f16 B-fragments
#pragma unroll
  for (int i = 0; i < 8; ++i) {
    int idx = tid + i * 256;
    int r   = idx >> 6;               // k row 0..31
    int d0  = (idx & 63) << 2;
    float4 v = *(const float4*)&ofraw[r][d0];
    const float vv[4] = {v.x, v.y, v.z, v.w};
#pragma unroll
    for (int j = 0; j < 4; ++j) {
      int d = d0 + j;
      ofb[d >> 4][(d & 15) + (r & 16)][r & 15] = (_Float16)vv[j];
    }
  }
#else
  // stage OF[b] (32 x 256 f32) once, as f16 B-fragments (direct global path)
#pragma unroll
  for (int i = 0; i < 8; ++i) {
    int idx = tid + i * 256;
    int r   = idx >> 6;               // k row 0..31
    int d0  = (idx & 63) << 2;
    float4 v = *(const float4*)(OF + ((size_t)b * KEIG + r) * D_MODEL + d0);
    const float vv[4] = {v.x, v.y, v.z, v.w};
#pragma unroll
    for (int j = 0; j < 4; ++j) {
      int d = d0 + j;
      ofb[d >> 4][(d & 15) + (r & 16)][r & 15] = (_Float16)vv[j];
    }
  }
#endif
  __syncthreads();

  const size_t obase = (size_t)b * N * D_MODEL;
  for (int base = blockIdx.x * 128; base < N; base += gridDim.x * 128) {
    const int row0 = base + wave * 16;
    const int m = lane & 15;
    const int n = row0 + m;

    v16h a;
    if (n < N) {
      const float4* er = (const float4*)(E + (size_t)n * KEIG);
      int off = (lane & 16) ? 4 : 0;  // lanes 0-15: K 0-15; 16-31: K 16-31
      float4 a0 = er[off], a1 = er[off + 1], a2 = er[off + 2], a3 = er[off + 3];
      a[0] = (_Float16)a0.x; a[1] = (_Float16)a0.y;
      a[2] = (_Float16)a0.z; a[3] = (_Float16)a0.w;
      a[4] = (_Float16)a1.x; a[5] = (_Float16)a1.y;
      a[6] = (_Float16)a1.z; a[7] = (_Float16)a1.w;
      a[8]  = (_Float16)a2.x; a[9]  = (_Float16)a2.y;
      a[10] = (_Float16)a2.z; a[11] = (_Float16)a2.w;
      a[12] = (_Float16)a3.x; a[13] = (_Float16)a3.y;
      a[14] = (_Float16)a3.z; a[15] = (_Float16)a3.w;
    } else {
#pragma unroll
      for (int e = 0; e < 16; ++e) a[e] = (_Float16)0.0f;
    }

#pragma unroll
    for (int t = 0; t < 16; ++t) {
      v16h bf = *(const v16h*)&ofb[t][lane][0];
      v8f d = {};
      d = __builtin_amdgcn_wmma_f32_16x16x32_f16(
          false, a, false, bf, (short)0, d, false, false);
#pragma unroll
      for (int v = 0; v < 8; ++v) {
        int row = row0 + v + ((lane & 16) ? 8 : 0);
        if (row < N)
          out[obase + (size_t)row * D_MODEL + (t << 4) + (lane & 15)] = d[v];
      }
    }
  }
}

// ---------------------------------------------------------------------------
extern "C" void kernel_launch(void* const* d_in, const int* in_sizes, int n_in,
                              void* d_out, int out_size, void* d_ws,
                              size_t ws_size, hipStream_t stream) {
  const float* x  = (const float*)d_in[0];
  const float* E  = (const float*)d_in[1];
  const float* ev = (const float*)d_in[2];
  const float* Wq = (const float*)d_in[3];
  const float* bq = (const float*)d_in[4];
  const float* Wk = (const float*)d_in[5];
  const float* bk = (const float*)d_in[6];
  const float* Wv = (const float*)d_in[7];
  const float* bv = (const float*)d_in[8];
  const float* bb = (const float*)d_in[9];
  const float* fw = (const float*)d_in[10];

  const int N = in_sizes[1] / KEIG;                  // 50000
  const int B = in_sizes[0] / (N * D_MODEL);         // 2

  float* wsf = (float*)d_ws;
  float* XF = wsf;                                   // B*32*256 f32
  float* S  = wsf + (size_t)B * KEIG * D_MODEL;      // 32 f32 (+pad)
  float* OF = S + 64;                                // B*32*256 f32

  k_init_ws<<<64, 256, 0, stream>>>(wsf, B * KEIG * D_MODEL + 64);
  k_fold<<<dim3(256, B), 256, 0, stream>>>(x, E, XF, S, N);
  k_spectral_attn<<<B * NHEAD, 256, 0, stream>>>(XF, S, ev, Wq, bq, Wk, bk,
                                                 Wv, bv, bb, fw, OF);
  k_unfold<<<dim3(256, B), 256, 0, stream>>>(OF, E, (float*)d_out, N);
}
